// GraphAttentionLayer_33998961115706
// MI455X (gfx1250) — compile-verified, matrix operand order checked by reading, not run on hardware
//
#include <hip/hip_runtime.h>
#include <hip/hip_bf16.h>

// GAT layer, B=2048 batches, N=24 nodes, F=64 features, H=F/2=32.
// One workgroup (256 threads = 8 wave32) per batch. WMMA f32<-f16 GEMMs.

typedef __attribute__((ext_vector_type(16))) _Float16 v16h;
typedef __attribute__((ext_vector_type(8)))  _Float16 v8h;
typedef __attribute__((ext_vector_type(8)))  float    v8f;
typedef __attribute__((ext_vector_type(4)))  float    v4f;

static __device__ __forceinline__ v16h cat8(v8h lo, v8h hi) {
    return __builtin_shufflevector(lo, hi, 0,1,2,3,4,5,6,7,8,9,10,11,12,13,14,15);
}

static __device__ __forceinline__ v8f wmma_f16(v16h a, v16h b, v8f c) {
    // (neg_a, A, neg_b, B, c_mod, C, reuse_a, reuse_b)
    return __builtin_amdgcn_wmma_f32_16x16x32_f16(false, a, false, b, (short)0, c, false, false);
}

#define GAT_N   24
#define GAT_F   64
#define GAT_H   32      // F/2
#define GAT_K   768     // N * H, contraction length of the attention GEMM
#define LRELU_A 0.01f
#define NEG_INF -1.0e12f

// ---- dynamic LDS layout (bytes), all offsets 16B aligned ----
#define OFF_AT   0          // _Float16 [24][768]  a^T ([n][k]); rows n>=24 over-read into
                            //   the arrays below (harmless: pollutes only discarded D cols)
#define OFF_WT   36864      // _Float16 [32][128]  W^T  ([d][f])
#define OFF_H    45056      // _Float16 [32][64]   h row-major (rows 24-31 zeroed)
#define OFF_HT   49152      // _Float16 [64][32]   h^T ([f][i], cols 24-31 zeroed: K-pad!)
#define OFF_U    53248      // float    [32][36]   U = h@W1   (stride 36: 144B rows, 16B aligned)
#define OFF_V    57856      // float    [32][36]   V = h@W2
#define OFF_EALL 62464      // float    [4][32][32] e partials (one per K-quarter)
#define OFF_ATT  78848      // _Float16 [32][32]   softmax(att), fully zero-padded
#define SMEM_BYTES 80896

__launch_bounds__(256)
__global__ void gat_wmma_kernel(const float* __restrict__ inp,
                                const int*   __restrict__ adj,
                                const float* __restrict__ Wg,
                                const float* __restrict__ ag,
                                float*       __restrict__ out)
{
    extern __shared__ char smem[];
    _Float16* sAT  = (_Float16*)(smem + OFF_AT);   // stride 768
    _Float16* sWT  = (_Float16*)(smem + OFF_WT);   // stride 128
    _Float16* sH   = (_Float16*)(smem + OFF_H);    // stride 64
    _Float16* sHT  = (_Float16*)(smem + OFF_HT);   // stride 32
    float*    sU   = (float*)   (smem + OFF_U);    // stride 36
    float*    sV   = (float*)   (smem + OFF_V);    // stride 36
    float*    sEal = (float*)   (smem + OFF_EALL); // 4 x [32][32]
    _Float16* sAtt = (_Float16*)(smem + OFF_ATT);  // stride 32

    const int b    = blockIdx.x;
    const int tid  = threadIdx.x;
    const int w    = tid >> 5;        // wave id 0..7 (wave32)
    const int lane = tid & 31;
    const int lm   = lane & 15;
    const int hi   = lane >> 4;       // K-half selector per WMMA fragment layout
    const int d0   = hi * 8;          // per-lane K base within a 32-chunk

    // ---------------- Phase A: stage operands into LDS (f32 -> f16) ----------------
    {
        const float* hB = inp + (size_t)b * (GAT_N * GAT_F);
        for (int i = tid; i < GAT_N * GAT_F; i += 256) {
            const int r = i >> 6, c = i & 63;
            const _Float16 hv = (_Float16)hB[i];
            sH [r * 64 + c] = hv;
            sHT[c * 32 + r] = hv;
        }
        // zero-pad: h rows 24..31 (A-tile rows), h^T cols 24..31 (K-pad, MUST be 0)
        for (int i = tid; i < 8 * 64; i += 256) sH [(24 + (i >> 6)) * 64 + (i & 63)] = (_Float16)0.f;
        for (int i = tid; i < 64 * 8; i += 256) sHT[(i >> 3) * 32 + 24 + (i & 7)]   = (_Float16)0.f;
        // att tile fully zeroed (K-pad cols 24..31 and rows 24..31 must be 0)
        for (int i = tid; i < 32 * 32; i += 256) sAtt[i] = (_Float16)0.f;

        const float* wB = Wg + (size_t)b * (2 * GAT_F * GAT_H);   // [128][32]
        for (int i = tid; i < 2 * GAT_F * GAT_H; i += 256) {
            const int f = i >> 5, d = i & 31;
            sWT[d * 128 + f] = (_Float16)wB[i];                    // W^T: [d][f]
        }
        // a: [768][24] -> a^T: [n][k]
        for (int i = tid; i < GAT_K * GAT_N; i += 256) {
            const int k = i / GAT_N, n = i - k * GAT_N;
            sAT[n * GAT_K + k] = (_Float16)ag[i];
        }
    }
    __syncthreads();

    // ---------------- Phase B: U = h@W[0:64,:], V = h@W[64:128,:]  (8 WMMA tiles) ----
    {
        const int matsel = w >> 2, mt = (w >> 1) & 1, nt = w & 1;
        const int m = mt * 16 + lm;               // A row (h row)
        const int n = nt * 16 + lm;               // B col (d index)
        const _Float16* aRow = sH  + m * 64;
        const _Float16* bRow = sWT + n * 128 + matsel * 64;
        v8f acc = {0.f,0.f,0.f,0.f,0.f,0.f,0.f,0.f};
#pragma unroll
        for (int k2 = 0; k2 < 2; ++k2) {
            const int ko = k2 * 32;
            v16h af = cat8(*(const v8h*)(aRow + ko + d0),
                           *(const v8h*)(aRow + ko + 16 + d0));
            v16h bf = cat8(*(const v8h*)(bRow + ko + d0),
                           *(const v8h*)(bRow + ko + 16 + d0));
            acc = wmma_f16(af, bf, acc);
        }
        float* dst = matsel ? sV : sU;
#pragma unroll
        for (int r = 0; r < 8; ++r)
            dst[(mt * 16 + r + 8 * hi) * 36 + nt * 16 + lm] = acc[r];
    }
    __syncthreads();

    // ---------------- Phase C: e = LReLU(U[i]+V[j]) @ a   (M=24,K=768,N=24) ----------
    // Jobs = (Mtile x 4-way K-split); each wave builds the A fragment ONCE per K-step
    // and feeds BOTH N-tiles with it (two back-to-back WMMAs sharing SRC0).
    {
        const int mt = w >> 2;                    // 0..1
        const int kh = w & 3;                     // K-quarter 0..3 (6 j-steps each)
        const int m  = mt * 16 + lm;
        const float* uRow = sU + m * 36;
        const v4f u0 = *(const v4f*)(uRow + d0);
        const v4f u1 = *(const v4f*)(uRow + d0 + 4);
        const v4f u2 = *(const v4f*)(uRow + 16 + d0);
        const v4f u3 = *(const v4f*)(uRow + 16 + d0 + 4);
        const _Float16* bBase0 = sAT + lm * GAT_K;          // nt=0: n = lm
        const _Float16* bBase1 = sAT + (16 + lm) * GAT_K;   // nt=1: n = 16+lm (n>=24 over-read OK)
        v8f acc0 = {0.f,0.f,0.f,0.f,0.f,0.f,0.f,0.f};
        v8f acc1 = {0.f,0.f,0.f,0.f,0.f,0.f,0.f,0.f};
        for (int s = 0; s < 6; ++s) {
            const int j = kh * 6 + s;             // K-block index == neighbor j
            const float* vRow = sV + j * 36;
            const v4f w0 = *(const v4f*)(vRow + d0);
            const v4f w1 = *(const v4f*)(vRow + d0 + 4);
            const v4f w2 = *(const v4f*)(vRow + 16 + d0);
            const v4f w3 = *(const v4f*)(vRow + 16 + d0 + 4);
            v16h af;
#pragma unroll
            for (int e = 0; e < 4; ++e) {
                const float t0 = u0[e] + w0[e];
                const float t1 = u1[e] + w1[e];
                const float t2 = u2[e] + w2[e];
                const float t3 = u3[e] + w3[e];
                af[e]      = (_Float16)fmaxf(t0, LRELU_A * t0);   // LeakyReLU
                af[e + 4]  = (_Float16)fmaxf(t1, LRELU_A * t1);
                af[e + 8]  = (_Float16)fmaxf(t2, LRELU_A * t2);
                af[e + 12] = (_Float16)fmaxf(t3, LRELU_A * t3);
            }
            const _Float16* bRow0 = bBase0 + j * 32;
            const _Float16* bRow1 = bBase1 + j * 32;
            v16h bf0 = cat8(*(const v8h*)(bRow0 + d0), *(const v8h*)(bRow0 + 16 + d0));
            v16h bf1 = cat8(*(const v8h*)(bRow1 + d0), *(const v8h*)(bRow1 + 16 + d0));
            acc0 = wmma_f16(af, bf0, acc0);
            acc1 = wmma_f16(af, bf1, acc1);
        }
        float* dst = sEal + kh * 1024;            // partial buffer for this K-quarter
#pragma unroll
        for (int r = 0; r < 8; ++r) {
            const int row = mt * 16 + r + 8 * hi;
            dst[row * 32 + lm]      = acc0[r];
            dst[row * 32 + 16 + lm] = acc1[r];
        }
    }
    __syncthreads();

    // ---------------- Phase D: 4-way reduce + mask + column softmax (dim=1) ----------
    if (tid < GAT_N) {
        const int j = tid;
        const int* adjB = adj + (size_t)b * (GAT_N * GAT_N);
        float mx = -3.0e38f;
        for (int i = 0; i < GAT_N; ++i) {
            const int o = i * 32 + j;
            float v = (sEal[o] + sEal[1024 + o]) + (sEal[2048 + o] + sEal[3072 + o]);
            v = (adjB[i * GAT_N + j] > 0) ? v : NEG_INF;
            sEal[o] = v;                           // reuse buffer 0 for masked e
            mx = fmaxf(mx, v);
        }
        float sum = 0.f;
        for (int i = 0; i < GAT_N; ++i) sum += __expf(sEal[i * 32 + j] - mx);
        const float inv = 1.0f / sum;
        for (int i = 0; i < GAT_N; ++i)
            sAtt[i * 32 + j] = (_Float16)(__expf(sEal[i * 32 + j] - mx) * inv);
    }
    __syncthreads();

    // ---------------- Phase E: out = ELU(att @ h)   (8 WMMA tiles, fused store) ------
    {
        const int mt = w >> 2, nt = w & 3;
        const int m = mt * 16 + lm;
        const int n = nt * 16 + lm;               // output feature
        const _Float16* aRow = sAtt + m * 32;
        v16h af = cat8(*(const v8h*)(aRow + d0),
                       *(const v8h*)(aRow + 16 + d0));
        const _Float16* bRow = sHT + n * 32;      // h^T[f][k], k-pad zeroed
        v16h bf = cat8(*(const v8h*)(bRow + d0),
                       *(const v8h*)(bRow + 16 + d0));
        v8f acc = {0.f,0.f,0.f,0.f,0.f,0.f,0.f,0.f};
        acc = wmma_f16(af, bf, acc);

        float* outB = out + (size_t)b * (GAT_N * GAT_F);
#pragma unroll
        for (int r = 0; r < 8; ++r) {
            const int row = mt * 16 + r + 8 * hi;
            if (row < GAT_N) {
                const float x = acc[r];
                outB[row * GAT_F + nt * 16 + lm] = (x > 0.f) ? x : expm1f(x);  // ELU(alpha=1)
            }
        }
    }
}

extern "C" void kernel_launch(void* const* d_in, const int* in_sizes, int n_in,
                              void* d_out, int out_size, void* d_ws, size_t ws_size,
                              hipStream_t stream) {
    const float* inp = (const float*)d_in[0];   // [B,24,64]  f32
    const int*   adj = (const int*)  d_in[1];   // [B,24,24]  i32
    const float* Wg  = (const float*)d_in[2];   // [B,128,32] f32
    const float* ag  = (const float*)d_in[3];   // [768,24]   f32
    float*       out = (float*)d_out;           // [B,24,64]  f32

    const int batches = in_sizes[0] / (GAT_N * GAT_F);   // 2048
    dim3 grid(batches), block(256);
    gat_wmma_kernel<<<grid, block, SMEM_BYTES, stream>>>(inp, adj, Wg, ag, out);
}
